// QGRULayer_25589415150152
// MI455X (gfx1250) — matrix operand order repaired and checked by reading, not compiled, not run on hardware
//
#include <hip/hip_runtime.h>
#include <hip/hip_bf16.h>
#include <stdint.h>

// Problem dims (fixed by the reference)
#define BB 4
#define LL 4096
#define DD 1024
#define HH 2048
#define MM (BB * LL)   // 16384 rows

typedef __attribute__((ext_vector_type(16))) __bf16 v16bf;
typedef __attribute__((ext_vector_type(8)))  __bf16 v8bf;
typedef __attribute__((ext_vector_type(8)))  float  v8f;

// Padded B-tile row: 64 K-elems + 8 pad (keeps 16B segment alignment,
// breaks the 128B-stride LDS bank pattern across the 16 fragment columns)
#define BSTRIDE 72
#define BTILE_BYTES (64 * BSTRIDE * 2)   // 9216 B per 64x64 bf16 tile
#define BTILE_ELEMS (64 * BSTRIDE)       // in uint16 units

__device__ __forceinline__ uint16_t f2bf(float f) {
  uint32_t u = __float_as_uint(f);
  uint32_t r = u + 0x7FFFu + ((u >> 16) & 1u);  // round-to-nearest-even
  return (uint16_t)(r >> 16);
}

// CDNA5 async copy: 16B global -> LDS, tracked by ASYNCcnt.
// lds_off is a byte offset from the workgroup LDS base (== start of dynamic
// shared memory; these kernels declare no static __shared__).
__device__ __forceinline__ void async_copy_b128(uint32_t lds_off, const void* gptr) {
  asm volatile("global_load_async_to_lds_b128 %0, %1, off"
               :: "v"(lds_off), "v"(gptr)
               : "memory");
}
__device__ __forceinline__ void wait_async0() {
  asm volatile("s_wait_asynccnt 0x0" ::: "memory");
}

__device__ __forceinline__ v16bf cat8(v8bf lo, v8bf hi) {
  return __builtin_shufflevector(lo, hi,
      0, 1, 2, 3, 4, 5, 6, 7, 8, 9, 10, 11, 12, 13, 14, 15);
}

// ---------------- pack x: f32 -> bf16, 4 elements per thread ----------------
__global__ void pack_x_kernel(const float* __restrict__ x,
                              uint16_t* __restrict__ xb, int n4) {
  int i = blockIdx.x * blockDim.x + threadIdx.x;
  if (i >= n4) return;
  float4 v = ((const float4*)x)[i];
  ushort4 o;
  o.x = f2bf(v.x); o.y = f2bf(v.y); o.z = f2bf(v.z); o.w = f2bf(v.w);
  ((ushort4*)xb)[i] = o;
}

// ------------- transpose [R][C] f32 -> [C][R] bf16 (weight prep) ------------
__global__ void transpose_bf16_kernel(const float* __restrict__ in,
                                      uint16_t* __restrict__ out, int R, int C) {
  int i = blockIdx.x * blockDim.x + threadIdx.x;
  if (i >= R * C) return;
  int r = i / C, c = i % C;
  out[(size_t)c * R + r] = f2bf(in[i]);
}

// ---------------------------------------------------------------------------
// Fused gate GEMM with async double-buffered B staging and a chunk-level
// fragment pipeline:
//   gf = x@Wf, gi = x@Wi (bf16 WMMA, f32 accum), fused activations.
// Block: 8 waves, tile 128(M) x 64(N); wave tile 16x64 for both gemms.
// K pipeline: 64-wide stages (2 x 32-wide WMMA chunks). Stage s+1's B tiles
// are DMA'd to LDS (ASYNCcnt) during stage s's WMMAs. Within a stage, the
// ds loads for chunk c+1 are issued into a SECOND live fragment set before
// the chunk-c WMMAs execute, so LDS latency is hidden and the fragment sets
// cannot be register-coalesced.
//
// Fragment layouts as loaded here (wave32; contiguous k, same order for A and B):
//   A 16x32 bf16: lane holds row M=lane&15; elems[0..15]=K[h*16..+16),
//                 h=lane>>4
//   B 32x16 bf16: lane holds col N=lane&15; elems[0..15]=K[h*16..+16)
// ---------------------------------------------------------------------------
__global__ void __launch_bounds__(256)
gates_gemm_kernel(const uint16_t* __restrict__ xb,   // [MM][DD] bf16
                  const uint16_t* __restrict__ wft,  // [HH][DD] bf16 (Wf^T)
                  const uint16_t* __restrict__ wit,  // [HH][DD] bf16 (Wi^T)
                  const float* __restrict__ bfv,     // [HH]
                  const float* __restrict__ biv,     // [HH]
                  float* __restrict__ f_gate,        // [MM][HH]
                  float* __restrict__ ri)            // [MM][HH]
{
  extern __shared__ uint16_t smem[];   // 2 stages x (Wf tile + Wi tile)
  const int tid  = threadIdx.x;
  const int lane = tid & 31;
  const int wave = tid >> 5;
  const int n0 = blockIdx.x * 64;
  const int m0 = blockIdx.y * 128 + wave * 16;
  const int arow = m0 + (lane & 15);
  const int khA = (lane >> 4) * 8;
  const int khB = (lane >> 4) * 16;
  const int STAGE_BYTES = 2 * BTILE_BYTES;   // 18432
  const int STAGE_ELEMS = STAGE_BYTES / 2;
  const int NSTAGES = DD / 64;               // 16

  v8f gf[4] = {};
  v8f gi[4] = {};
  const uint16_t* arowp = xb + (size_t)arow * DD;

  // copy stage s: 64 N-cols x 64 K of Wf^T and Wi^T slices
  auto copy_stage = [&](int s) {
    uint32_t base = (uint32_t)((s & 1) * STAGE_BYTES);
    int k0 = s * 64;
#pragma unroll
    for (int t = 0; t < 2; ++t) {
      int flat = tid + t * 256;          // 512 x 16B per tile
      int col = flat >> 3, seg = flat & 7;
      uint32_t loff = base + (uint32_t)col * (BSTRIDE * 2) + (uint32_t)seg * 16;
      size_t goff = (size_t)(n0 + col) * DD + k0 + seg * 8;
      async_copy_b128(loff, wft + goff);
      async_copy_b128(loff + BTILE_BYTES, wit + goff);
    }
  };

  // read the 8 B fragments (Wf + Wi) of one 32-wide K chunk from LDS
  auto load_frags = [&](uint32_t base_e, int c, v16bf (&f4)[4], v16bf (&i4)[4]) {
#pragma unroll
    for (int j = 0; j < 4; ++j) {
      uint32_t be = base_e + (uint32_t)(j * 16 + (lane & 15)) * BSTRIDE
                    + (uint32_t)(c * 32 + khB);
      f4[j] = cat8(*(const v8bf*)(smem + be),
                   *(const v8bf*)(smem + be + 8));
      i4[j] = cat8(*(const v8bf*)(smem + be + BTILE_ELEMS),
                   *(const v8bf*)(smem + be + BTILE_ELEMS + 8));
    }
  };

  auto wmma8 = [&](v16bf a, v16bf (&f4)[4], v16bf (&i4)[4]) {
#pragma unroll
    for (int j = 0; j < 4; ++j) {
      gf[j] = __builtin_amdgcn_wmma_f32_16x16x32_bf16(
          false, a, false, f4[j], (short)0, gf[j], false, false);
      gi[j] = __builtin_amdgcn_wmma_f32_16x16x32_bf16(
          false, a, false, i4[j], (short)0, gi[j], false, false);
    }
  };

  copy_stage(0);
  wait_async0();
  __syncthreads();

  v16bf curF[4], curI[4], nxtF[4], nxtI[4];
  load_frags(0, 0, curF, curI);              // stage 0, chunk 0

  for (int s = 0; s < NSTAGES; ++s) {
    if (s + 1 < NSTAGES) copy_stage(s + 1);  // DMA overlaps WMMAs below
    const uint32_t base_e = (uint32_t)((s & 1) * STAGE_ELEMS);
    const int k0 = s * 64;

    // chunk 0: prefetch chunk-1 fragments into the second live set first
    v16bf a0 = cat8(*(const v8bf*)((const __bf16*)(arowp + k0) + khB),
                    *(const v8bf*)((const __bf16*)(arowp + k0) + khB + 8));
    load_frags(base_e, 1, nxtF, nxtI);
    wmma8(a0, curF, curI);

    // chunk 1
    v16bf a1 = cat8(*(const v8bf*)((const __bf16*)(arowp + k0 + 32) + khB),
                    *(const v8bf*)((const __bf16*)(arowp + k0 + 32) + khB + 8));
    wmma8(a1, nxtF, nxtI);

    wait_async0();
    __syncthreads();
    if (s + 1 < NSTAGES)
      load_frags((uint32_t)(((s + 1) & 1) * STAGE_ELEMS), 0, curF, curI);
  }

  // Epilogue: row = m0 + r + 8*(lane>>4), col = n0 + 16j + (lane&15)
  const float invHm1 = 1.0f / (float)(HH - 1);
  const int rowoff = (lane >> 4) * 8;
#pragma unroll
  for (int j = 0; j < 4; ++j) {
    int col = n0 + j * 16 + (lane & 15);
    float bb = bfv[col];
    float ib = biv[col];
    float sc = (float)col * invHm1;     // linspace(0,1,H)[col]
#pragma unroll
    for (int r = 0; r < 8; ++r) {
      int row = m0 + rowoff + r;
      float gfe = gf[j][r] + bb;
      float gie = gi[j][r] + ib;
      float rem = sc / (1.0f + __expf(-gfe));   // sigmoid * scale
      size_t idx = (size_t)row * HH + col;
      f_gate[idx] = 1.0f - rem;
      ri[idx] = rem * tanhf(gie);
    }
  }
}

// --------------- sequential linear recurrence over L, per channel ----------
// h[t] = f[t]*h[t-1] + ri[t], h[-1] = hidden.  One thread per (b, channel),
// coalesced across H; global_prefetch 16 steps ahead hides HBM latency.
__global__ void scan_kernel(const float* __restrict__ f_gate,
                            const float* __restrict__ ri,
                            const float* __restrict__ hidden,
                            uint16_t* __restrict__ h_bf,  // [MM][HH] bf16
                            float* __restrict__ h_last)   // [BB][HH]
{
  int tid = blockIdx.x * blockDim.x + threadIdx.x;  // 0 .. BB*HH-1
  int b = tid / HH;
  int c = tid % HH;
  float h = hidden[tid];
  size_t idx = (size_t)b * LL * HH + c;
  for (int t = 0; t < LL; ++t, idx += HH) {
    if (t + 16 < LL) {
      __builtin_prefetch(f_gate + idx + 16 * HH, 0, 0);
      __builtin_prefetch(ri + idx + 16 * HH, 0, 0);
    }
    float fg = f_gate[idx];
    float rv = ri[idx];
    h = fmaf(fg, h, rv);
    h_bf[idx] = f2bf(h);
  }
  h_last[tid] = h;
}

// --------------------- output GEMM: y = h @ Wo + bo ------------------------
// Same async double-buffered B staging + chunk-level fragment pipeline.
__global__ void __launch_bounds__(256)
out_gemm_kernel(const uint16_t* __restrict__ hb,   // [MM][HH] bf16
                const uint16_t* __restrict__ wot,  // [DD][HH] bf16 (Wo^T)
                const float* __restrict__ bo,      // [DD]
                float* __restrict__ y)             // [MM][DD] f32
{
  extern __shared__ uint16_t smem[];   // 2 stages x Wo tile
  const int tid  = threadIdx.x;
  const int lane = tid & 31;
  const int wave = tid >> 5;
  const int n0 = blockIdx.x * 64;
  const int m0 = blockIdx.y * 128 + wave * 16;
  const int arow = m0 + (lane & 15);
  const int khA = (lane >> 4) * 8;
  const int khB = (lane >> 4) * 16;
  const int NSTAGES = HH / 64;               // 32

  v8f acc[4] = {};
  const uint16_t* arowp = hb + (size_t)arow * HH;

  auto copy_stage = [&](int s) {
    uint32_t base = (uint32_t)((s & 1) * BTILE_BYTES);
    int k0 = s * 64;
#pragma unroll
    for (int t = 0; t < 2; ++t) {
      int flat = tid + t * 256;
      int col = flat >> 3, seg = flat & 7;
      uint32_t loff = base + (uint32_t)col * (BSTRIDE * 2) + (uint32_t)seg * 16;
      async_copy_b128(loff, wot + (size_t)(n0 + col) * HH + k0 + seg * 8);
    }
  };

  auto load_frags = [&](uint32_t base_e, int c, v16bf (&b4)[4]) {
#pragma unroll
    for (int j = 0; j < 4; ++j) {
      uint32_t be = base_e + (uint32_t)(j * 16 + (lane & 15)) * BSTRIDE
                    + (uint32_t)(c * 32 + khB);
      b4[j] = cat8(*(const v8bf*)(smem + be),
                   *(const v8bf*)(smem + be + 8));
    }
  };

  auto wmma4 = [&](v16bf a, v16bf (&b4)[4]) {
#pragma unroll
    for (int j = 0; j < 4; ++j) {
      acc[j] = __builtin_amdgcn_wmma_f32_16x16x32_bf16(
          false, a, false, b4[j], (short)0, acc[j], false, false);
    }
  };

  copy_stage(0);
  wait_async0();
  __syncthreads();

  v16bf cur[4], nxt[4];
  load_frags(0, 0, cur);

  for (int s = 0; s < NSTAGES; ++s) {
    if (s + 1 < NSTAGES) copy_stage(s + 1);
    const uint32_t base_e = (uint32_t)((s & 1) * (BTILE_BYTES / 2));
    const int k0 = s * 64;

    v16bf a0 = cat8(*(const v8bf*)((const __bf16*)(arowp + k0) + khB),
                    *(const v8bf*)((const __bf16*)(arowp + k0) + khB + 8));
    load_frags(base_e, 1, nxt);
    wmma4(a0, cur);

    v16bf a1 = cat8(*(const v8bf*)((const __bf16*)(arowp + k0 + 32) + khB),
                    *(const v8bf*)((const __bf16*)(arowp + k0 + 32) + khB + 8));
    wmma4(a1, nxt);

    wait_async0();
    __syncthreads();
    if (s + 1 < NSTAGES)
      load_frags((uint32_t)(((s + 1) & 1) * (BTILE_BYTES / 2)), 0, cur);
  }

  const int rowoff = (lane >> 4) * 8;
#pragma unroll
  for (int j = 0; j < 4; ++j) {
    int col = n0 + j * 16 + (lane & 15);
    float bb = bo[col];
#pragma unroll
    for (int r = 0; r < 8; ++r) {
      int row = m0 + rowoff + r;
      y[(size_t)row * DD + col] = acc[j][r] + bb;
    }
  }
}

extern "C" void kernel_launch(void* const* d_in, const int* in_sizes, int n_in,
                              void* d_out, int out_size, void* d_ws, size_t ws_size,
                              hipStream_t stream) {
  (void)in_sizes; (void)n_in; (void)out_size; (void)ws_size;
  const float* x      = (const float*)d_in[0];  // [B,L,D]
  const float* hidden = (const float*)d_in[1];  // [B,H]
  const float* Wf     = (const float*)d_in[2];  // [D,H]
  const float* bf     = (const float*)d_in[3];  // [H]
  const float* Wi     = (const float*)d_in[4];  // [D,H]
  const float* bi     = (const float*)d_in[5];  // [H]
  const float* Wo     = (const float*)d_in[6];  // [H,D]
  const float* bo     = (const float*)d_in[7];  // [D]

  float* y      = (float*)d_out;                 // [B,L,D]
  float* h_last = y + (size_t)MM * DD;           // [B,H] (tuple tail)

  uint8_t* ws = (uint8_t*)d_ws;
  size_t off = 0;
  auto carve = [&](size_t bytes) -> uint8_t* {
    uint8_t* p = ws + off;
    off = (off + bytes + 255) & ~(size_t)255;
    return p;
  };
  uint16_t* xb     = (uint16_t*)carve((size_t)MM * DD * 2);  // x bf16
  uint16_t* wft    = (uint16_t*)carve((size_t)HH * DD * 2);  // Wf^T bf16
  uint16_t* wit    = (uint16_t*)carve((size_t)HH * DD * 2);  // Wi^T bf16
  uint16_t* wot    = (uint16_t*)carve((size_t)DD * HH * 2);  // Wo^T bf16
  float*    f_gate = (float*)carve((size_t)MM * HH * 4);     // forget f32
  float*    rri    = (float*)carve((size_t)MM * HH * 4);     // remember*inp f32
  uint16_t* hb     = (uint16_t*)carve((size_t)MM * HH * 2);  // h bf16

  // 1) precision/layout prep
  pack_x_kernel<<<(MM * DD / 4 + 255) / 256, 256, 0, stream>>>(x, xb, MM * DD / 4);
  transpose_bf16_kernel<<<(DD * HH + 255) / 256, 256, 0, stream>>>(Wf, wft, DD, HH);
  transpose_bf16_kernel<<<(DD * HH + 255) / 256, 256, 0, stream>>>(Wi, wit, DD, HH);
  transpose_bf16_kernel<<<(HH * DD + 255) / 256, 256, 0, stream>>>(Wo, wot, HH, DD);

  // 2) fused gate GEMMs + activations (36KB dynamic LDS: 2 stages x 2 tiles)
  gates_gemm_kernel<<<dim3(HH / 64, MM / 128), 256, 4 * BTILE_BYTES, stream>>>(
      xb, wft, wit, bf, bi, f_gate, rri);

  // 3) recurrence scan (8192 independent channels)
  scan_kernel<<<(BB * HH) / 256, 256, 0, stream>>>(f_gate, rri, hidden, hb, h_last);

  // 4) output GEMM + bias (18KB dynamic LDS)
  out_gemm_kernel<<<dim3(DD / 64, MM / 128), 256, 2 * BTILE_BYTES, stream>>>(
      hb, wot, bo, y);
}
